// GNNLayer_3736621548214
// MI455X (gfx1250) — matrix operand order repaired and checked by reading, not compiled, not run on hardware
//
#include <hip/hip_runtime.h>
#include <hip/hip_bf16.h>

// ---------------------------------------------------------------------------
// GNN layer for MI455X (gfx1250, wave32, WMMA + async LDS copies).
// Sizes fixed by the reference: B=16, N=64, D=256, E=256.
//   pairs M = B*N*N = 65536, K_edge = 256, D = 256.
// Factorization: h = relu(U[bi] + V[bj] + edges @ W_in[512:768] + b_in),
//   U = nodes @ W_in[0:256], V = nodes @ W_in[256:512].
// GEMMs: v_wmma_f32_16x16x32_bf16; weight tiles pre-swizzled to the exact
// B-fragment layout so tile staging is global_load_async_to_lds_b128.
// ---------------------------------------------------------------------------

typedef __attribute__((ext_vector_type(16))) __bf16 v16bf;
typedef __attribute__((ext_vector_type(8)))  float  v8f;

union Frag { v16bf v; unsigned u[8]; uint4 q[2]; };

__device__ __forceinline__ unsigned short f2bf(float x) {
    unsigned u = __float_as_uint(x);
    u += 0x7fffu + ((u >> 16) & 1u);           // round-to-nearest-even
    return (unsigned short)(u >> 16);
}

__device__ __forceinline__ unsigned pack2bf(float a, float b) {
    unsigned ua = __float_as_uint(a), ub = __float_as_uint(b);
    ua += 0x7fffu + ((ua >> 16) & 1u);
    ub += 0x7fffu + ((ub >> 16) & 1u);
    return (ua >> 16) | (ub & 0xffff0000u);    // lo = bf16(a), hi = bf16(b)
}

// async 16B copy global -> LDS (VDST = per-lane LDS byte address).
// Tracked by ASYNCcnt; waited with s_wait_asynccnt before the barrier.
__device__ __forceinline__ void async_ld_b128(unsigned lds,
                                              unsigned long long gaddr) {
    asm volatile("global_load_async_to_lds_b128 %0, %1, off"
                 :: "v"(lds), "v"(gaddr) : "memory");
}
__device__ __forceinline__ void wait_async0() {
    asm volatile("s_wait_asynccnt 0x0" ::: "memory");
}

// ---------------------------------------------------------------------------
// Weight swizzle: B[K][256] (f32) -> fragment-major packed bf16 dwords:
//   dword o = (kc*4 + nb)*1024 + (t*32 + lane)*8 + i
//     element: k = kc*32 + 2*(i + (lane>>4)*8) (+1 in hi16),
//              n = nb*64 + t*16 + (lane&15)
// Lane-contiguous k order (k = 16*half + s) matching the GEMM's A fragments, so the
// GEMM's B staging is a pure 16B-chunk copy.
// ---------------------------------------------------------------------------
__device__ __forceinline__ unsigned swz_pack(const float* __restrict__ W,
                                             int o) {
    const int blk = o >> 10, rem = o & 1023;
    const int tl = rem >> 3, i = rem & 7;
    const int t = tl >> 5, lane = tl & 31;
    const int half = lane >> 4, nrow = lane & 15;
    const int k = (blk >> 2) * 32 + 2 * (i + half * 8);
    const int n = (blk & 3) * 64 + t * 16 + nrow;
    return pack2bf(W[(size_t)k * 256 + n], W[(size_t)(k + 1) * 256 + n]);
}

// prep: nodes -> bf16 rowmajor; W_in (768x256), W_out (256x256) -> swizzled
__global__ __launch_bounds__(256) void prep_kernel(
    const float* __restrict__ W_in, const float* __restrict__ W_out,
    const float* __restrict__ nodes, unsigned* __restrict__ WinSwz,
    unsigned* __restrict__ WoutSwz, unsigned short* __restrict__ nodesbf) {
    int t = blockIdx.x * blockDim.x + threadIdx.x;      // 0 .. 262143
    nodesbf[t] = f2bf(nodes[t]);
    if (t < 98304) WinSwz[t]  = swz_pack(W_in, t);      // 768*256/2 dwords
    if (t < 32768) WoutSwz[t] = swz_pack(W_out, t);     // 256*256/2 dwords
}

// ---------------------------------------------------------------------------
// Tiled WMMA GEMM: C[M,NT=256] = A[M,K] @ B[K,256] (+ epilogue).
// Block = 128 threads = 4 waves. Block tile 64(M) x 64(N), K step 32.
// Wave w computes rows [16w,16w+16) x 64 cols = 4 accum tiles.
// Double-buffered LDS (pipeline unrolled x2 so buffer addresses are
// literals); B tile (and A tile for bf16-A modes) staged with
// global_load_async_to_lds_b128; one barrier + one asynccnt wait per K-step.
// B fragments software-pipelined in 4 distinct register sets so ds waits
// overlap WMMA issue. Requires K % 64 == 0 (K = 256 here).
//
// MODE 0: Cout[m,n] = acc                                (A = bf16 rowmajor)
// MODE 1: p=m; bi=p>>6; b=bi>>6; j=p&63;
//         h = relu(acc + U[bi,n] + V[b*64+j,n] + bias[n])  (A = f32 rowmajor)
// MODE 2: Cout[m,n] = addsrc[m,n] + relu(acc + bias[n])   (A = bf16 rowmajor)
// ---------------------------------------------------------------------------
template <int MODE>
__global__ __launch_bounds__(128) void wmma_gemm(
    const void* __restrict__ Aptr, const unsigned* __restrict__ Bswz,
    float* __restrict__ Cout,
    const float* __restrict__ U, const float* __restrict__ Vv,
    const float* __restrict__ bias, const float* __restrict__ addsrc,
    int M, int K, int NT) {
    __shared__ unsigned As[2][64 * 20];       // 64 rows x 16 dwords, pad 20
    __shared__ unsigned Bsw[2][4 * 32 * 12];  // 4 tiles x 32 lanes x 8, pad 12

    const int tid     = threadIdx.x;
    const int block_m = blockIdx.x * 64;
    const int nb      = blockIdx.y;           // 64-col block (0..3)
    const int block_n = nb * 64;
    const int w    = tid >> 5;
    const int lane = tid & 31;
    const int half = lane >> 4;
    const int lrow = lane & 15;

    const unsigned asBase  = (unsigned)(size_t)(void*)&As[0][0];
    const unsigned bswBase = (unsigned)(size_t)(void*)&Bsw[0][0];

    // MODE 1 A staging roles (f32 -> bf16 conversion in registers)
    const int arow = tid >> 1, ach = tid & 1;
    unsigned ta[8];

    v8f acc[4];
#pragma unroll
    for (int t = 0; t < 4; ++t)
#pragma unroll
        for (int r = 0; r < 8; ++r) acc[t][r] = 0.0f;

    auto issueAsyncB = [&](int kb, int buf) {
        // tile = 1024 dwords = 256 x 16B chunks; thread: chunks 2tid, 2tid+1
        const char* Bb = (const char*)(Bswz + ((size_t)((kb >> 5) * 4 + nb)
                                               << 10));
#pragma unroll
        for (int s = 0; s < 2; ++s) {
            const int c = tid * 2 + s;
            const int tl = c >> 1, ih = (c & 1) * 4;
            async_ld_b128(bswBase + (unsigned)(buf * 6144 + (tl * 12 + ih) * 4),
                          (unsigned long long)(size_t)(Bb + (size_t)c * 16));
        }
    };
    auto issueAsyncA = [&](int kb, int buf) {     // bf16-A modes: pure copy
        const char* Ab = (const char*)Aptr;
#pragma unroll
        for (int s = 0; s < 2; ++s) {
            const int c = tid * 2 + s;
            const int row = c >> 2, q = c & 3;
            const char* g = Ab + (((size_t)(block_m + row) * (K >> 1) +
                                   (kb >> 1)) << 2) + q * 16;
            async_ld_b128(asBase + (unsigned)(buf * 5120 + row * 80 + q * 16),
                          (unsigned long long)(size_t)g);
        }
    };
    auto loadAregs = [&](int kb) {                // MODE 1: edges f32 -> bf16
        const float* Af = (const float*)Aptr +
                          (size_t)(block_m + arow) * K + kb + ach * 16;
#pragma unroll
        for (int j = 0; j < 4; ++j) {
            float4 f = ((const float4*)Af)[j];
            ta[2 * j]     = pack2bf(f.x, f.y);
            ta[2 * j + 1] = pack2bf(f.z, f.w);
        }
    };
    auto storeAregs = [&](int buf) {
        unsigned* da = &As[buf][arow * 20 + ach * 8];
#pragma unroll
        for (int j = 0; j < 8; ++j) da[j] = ta[j];    // 2x ds_store_b128
    };

    const int niter = K >> 5;                   // even (K % 64 == 0)
    issueAsyncB(0, 0);
    if (MODE == 1) {
        loadAregs(0);
        storeAregs(0);
    } else {
        issueAsyncA(0, 0);
    }

    auto pipeStep = [&](int i, int buf) {       // buf is a literal at call site
        wait_async0();
        __syncthreads();                 // tile i visible; buffers reusable
        const int more = (i + 1 < niter);
        if (more) {                      // prefetch tile i+1 into other buffer
            issueAsyncB((i + 1) << 5, buf ^ 1);
            if (MODE == 1) loadAregs((i + 1) << 5);
            else           issueAsyncA((i + 1) << 5, buf ^ 1);
        }
        const unsigned* AsB = &As[buf][0];
        const unsigned* BsB = &Bsw[buf][0];
        // software-pipelined fragments: 4 distinct register sets, loads of
        // fragment t+1 issued before WMMA t so ds waits overlap WMMA issue
        Frag af, b0, b1, b2, b3;
        af.q[0] = *(const uint4*)&AsB[(w * 16 + lrow) * 20 + half * 8];
        af.q[1] = *(const uint4*)&AsB[(w * 16 + lrow) * 20 + half * 8 + 4];
        const uint4* p0 = (const uint4*)&BsB[(0 * 32 + lane) * 12];
        b0.q[0] = p0[0]; b0.q[1] = p0[1];
        const uint4* p1 = (const uint4*)&BsB[(1 * 32 + lane) * 12];
        b1.q[0] = p1[0]; b1.q[1] = p1[1];
        acc[0] = __builtin_amdgcn_wmma_f32_16x16x32_bf16(
            false, af.v, false, b0.v, (short)0, acc[0], false, false);
        const uint4* p2 = (const uint4*)&BsB[(2 * 32 + lane) * 12];
        b2.q[0] = p2[0]; b2.q[1] = p2[1];
        acc[1] = __builtin_amdgcn_wmma_f32_16x16x32_bf16(
            false, af.v, false, b1.v, (short)0, acc[1], false, false);
        const uint4* p3 = (const uint4*)&BsB[(3 * 32 + lane) * 12];
        b3.q[0] = p3[0]; b3.q[1] = p3[1];
        acc[2] = __builtin_amdgcn_wmma_f32_16x16x32_bf16(
            false, af.v, false, b2.v, (short)0, acc[2], false, false);
        acc[3] = __builtin_amdgcn_wmma_f32_16x16x32_bf16(
            false, af.v, false, b3.v, (short)0, acc[3], false, false);
        if (MODE == 1 && more) storeAregs(buf ^ 1);
    };

    for (int i = 0; i < niter; i += 2) {        // x2 unroll: literal buffers
        pipeStep(i, 0);
        pipeStep(i + 1, 1);
    }

    // ---- epilogue: C layout = VGPR r -> (M = r + half*8, N = lane&15) ----
#pragma unroll
    for (int t = 0; t < 4; ++t) {
        const int gn = block_n + t * 16 + lrow;
#pragma unroll
        for (int r = 0; r < 8; ++r) {
            const int gm = block_m + w * 16 + half * 8 + r;
            float c = acc[t][r];
            if (MODE == 0) {
                Cout[(size_t)gm * NT + gn] = c;
            } else if (MODE == 1) {
                const int p = gm, bi = p >> 6, b = bi >> 6, j = p & 63;
                float hv = c + U[(size_t)bi * 256 + gn] +
                           Vv[(size_t)(b * 64 + j) * 256 + gn] + bias[gn];
                Cout[(size_t)p * 256 + gn] = fmaxf(hv, 0.0f);
            } else {
                float o = fmaxf(c + bias[gn], 0.0f);
                Cout[(size_t)gm * NT + gn] = addsrc[(size_t)gm * NT + gn] + o;
            }
        }
    }
}

// ---------------------------------------------------------------------------
// softmax + weighted reduction. One block per (b,i) row: 1024 blocks x 256.
// coef[j] = h[bi,j,:] . W_coef + b_coef  (h already relu'd)
// attn = softmax_j(coef - 1e9*[j==i]);  residual[bi,:] = sum_j attn_j h[bi,j,:]
// h (64 MB) is L2-resident (192 MB L2), so the double pass over h is cheap.
// ---------------------------------------------------------------------------
__global__ __launch_bounds__(256) void softmax_residual(
    const float* __restrict__ h, const float* __restrict__ Wc,
    const float* __restrict__ bcoef, unsigned short* __restrict__ resbf) {
    __shared__ float cpart[256];
    __shared__ float coefs[64];
    __shared__ float attn[64];
    const int bi  = blockIdx.x;        // 0..1023  (bi = b*64 + i)
    const int tid = threadIdx.x;

    {   // coef partial dots: thread (j = tid/4, q = tid%4) covers 64 dims
        const int j = tid >> 2, q = tid & 3;
        const float* hr = h + ((size_t)(bi * 64 + j)) * 256 + q * 64;
        const float* wc = Wc + q * 64;
        float s = 0.0f;
#pragma unroll
        for (int n = 0; n < 64; n += 4) {
            float4 hv = *(const float4*)(hr + n);
            float4 wv = *(const float4*)(wc + n);
            s += hv.x * wv.x + hv.y * wv.y + hv.z * wv.z + hv.w * wv.w;
        }
        cpart[tid] = s;
    }
    __syncthreads();
    if (tid < 64) {
        float c = cpart[tid * 4] + cpart[tid * 4 + 1] + cpart[tid * 4 + 2] +
                  cpart[tid * 4 + 3] + bcoef[0];
        if (tid == (bi & 63)) c -= 1e9f;   // mask diagonal j == i
        coefs[tid] = c;
    }
    __syncthreads();
    if (tid == 0) {
        float m = -1e30f;
        for (int k = 0; k < 64; ++k) m = fmaxf(m, coefs[k]);
        float ss = 0.0f;
        for (int k = 0; k < 64; ++k) {
            float e = __expf(coefs[k] - m);
            attn[k] = e;
            ss += e;
        }
        float inv = 1.0f / ss;
        for (int k = 0; k < 64; ++k) attn[k] *= inv;
    }
    __syncthreads();
    {   // residual: thread tid = channel n, coalesced 1 KB per j step
        float r = 0.0f;
        const float* hb = h + ((size_t)bi * 64) * 256 + tid;
#pragma unroll 4
        for (int k = 0; k < 64; ++k) r += attn[k] * hb[(size_t)k * 256];
        resbf[(size_t)bi * 256 + tid] = f2bf(r);
    }
}

// ---------------------------------------------------------------------------
// launcher
// inputs: nodes, edges, W_in, b_in, W_coef, b_coef, W_out, b_out, B, N
// d_out:  new_nodes [1024*256] ++ h [65536*256]   (f32)
// ---------------------------------------------------------------------------
extern "C" void kernel_launch(void* const* d_in, const int* in_sizes, int n_in,
                              void* d_out, int out_size, void* d_ws,
                              size_t ws_size, hipStream_t stream) {
    const float* nodes  = (const float*)d_in[0];
    const float* edges  = (const float*)d_in[1];
    const float* W_in   = (const float*)d_in[2];
    const float* b_in   = (const float*)d_in[3];
    const float* W_coef = (const float*)d_in[4];
    const float* b_coef = (const float*)d_in[5];
    const float* W_out  = (const float*)d_in[6];
    const float* b_out  = (const float*)d_in[7];

    float* newn = (float*)d_out;          // [1024,256]
    float* hbuf = newn + 262144;          // [65536,256]

    // workspace carve-up (~3.5 MB total)
    float* U = (float*)d_ws;                                  // [1024,256] f32
    float* V = U + 262144;                                    // [1024,256] f32
    unsigned short* resbf   = (unsigned short*)(V + 262144);  // [1024,256] bf16
    unsigned short* nodesbf = resbf + 262144;                 // [1024,256] bf16
    unsigned* WinSwz  = (unsigned*)(nodesbf + 262144);        // 98304 dwords
    unsigned* WoutSwz = WinSwz + 98304;                       // 32768 dwords

    prep_kernel<<<1024, 256, 0, stream>>>(W_in, W_out, nodes, WinSwz, WoutSwz,
                                          nodesbf);

    dim3 blk(128);
    dim3 gSmall(16, 4);     // M=1024, NT=256
    // U = nodes @ W_in[0:256,:],  V = nodes @ W_in[256:512,:]
    wmma_gemm<0><<<gSmall, blk, 0, stream>>>(nodesbf, WinSwz, U, nullptr,
                                             nullptr, nullptr, nullptr, 1024,
                                             256, 256);
    wmma_gemm<0><<<gSmall, blk, 0, stream>>>(nodesbf, WinSwz + 8 * 4096, V,
                                             nullptr, nullptr, nullptr,
                                             nullptr, 1024, 256, 256);
    // h = relu(edges @ W_in[512:768,:] + U[bi] + V[bj] + b_in)
    dim3 gPair(1024, 4);    // M=65536, NT=256
    wmma_gemm<1><<<gPair, blk, 0, stream>>>(edges, WinSwz + 16 * 4096, hbuf, U,
                                            V, b_in, nullptr, 65536, 256, 256);

    softmax_residual<<<1024, 256, 0, stream>>>(hbuf, W_coef, b_coef, resbf);

    // new_nodes = nodes + relu(residual @ W_out + b_out)
    wmma_gemm<2><<<gSmall, blk, 0, stream>>>(resbf, WoutSwz, newn, nullptr,
                                             nullptr, b_out, nodes, 1024, 256,
                                             256);
}
